// MultiHeadAttention_60936995996260
// MI455X (gfx1250) — matrix operand order repaired and checked by reading, not compiled, not run on hardware
//
#include <hip/hip_runtime.h>

// ---------------------------------------------------------------------------
// MultiHeadAttention forward for MI455X (gfx1250), f16 WMMA + flash softmax.
// B=8, T=1024, D_MODEL=768, H=12, D_HEAD=64.
// Round 3: ping-pong (two named fragment sets) K-pipeline with #pragma unroll 1
// to kill the round-2 scratch spills; attention softmax fused to minimize
// live VGPRs; launch bounds on all WMMA kernels.
// ---------------------------------------------------------------------------

typedef __attribute__((ext_vector_type(16))) _Float16 v16h;
typedef __attribute__((ext_vector_type(8)))  _Float16 v8h;
typedef __attribute__((ext_vector_type(4)))  _Float16 v4h;
typedef __attribute__((ext_vector_type(8)))  float    v8f;

#define DM    768
#define HEADS 12
#define DHEAD 64
#define TSEQ  1024
#define BATCH 8
#define MROWS (BATCH * TSEQ)   // 8192

#define WMMA16(a, b, c) __builtin_amdgcn_wmma_f32_16x16x32_f16( \
    false, (a), false, (b), (short)0, (c), false, false)

// ---- fragment loaders (contiguous k = 16*(lane/16) + s order for both A and B, wave32) ---------------

__device__ __forceinline__ v16h make16(v8h lo, v8h hi) {
  v16h r;
#pragma unroll
  for (int i = 0; i < 8; ++i) { r[i] = lo[i]; r[i + 8] = hi[i]; }
  return r;
}

// A-matrix 16x32 f16: lane&15 = row; lanes 0-15 hold K0..15, lanes 16-31 hold
// K16..31 sequentially.  p = &A[row][kb] (K-contiguous).
__device__ __forceinline__ v16h load_afrag(const _Float16* p, int lane) {
  const int kh = (lane >> 4) * 16;
  v8h lo = *(const v8h*)(p + kh);
  v8h hi = *(const v8h*)(p + kh + 8);
  return make16(lo, hi);
}

// B-matrix 32x16 f16: lane&15 = col n; lanes 0-15 hold K0..15, lanes 16-31 hold
// K16..31 sequentially.  p = &Bt[n][kb] (K-contiguous, i.e. B stored N-major).
__device__ __forceinline__ v16h load_bfrag(const _Float16* p, int lane) {
  const int kh = (lane >> 4) * 16;
  v8h lo = *(const v8h*)(p + kh);
  v8h hi = *(const v8h*)(p + kh + 8);
  return make16(lo, hi);
}

// ---- shared 32x64-per-wave GEMM core, ping-pong K pipeline -----------------
// acc[mt][nt] += A[m0+mt*16 .. +15][0..K) * Bt[n0+nt*16 .. +15][0..K)^T

#define LOAD_SET(dst_a0, dst_a1, dst_b0, dst_b1, dst_b2, dst_b3, koff)   \
  do {                                                                   \
    dst_a0 = load_afrag(ar0 + (koff), lane);                             \
    dst_a1 = load_afrag(ar1 + (koff), lane);                             \
    dst_b0 = load_bfrag(bp0 + (koff), lane);                             \
    dst_b1 = load_bfrag(bp1 + (koff), lane);                             \
    dst_b2 = load_bfrag(bp2 + (koff), lane);                             \
    dst_b3 = load_bfrag(bp3 + (koff), lane);                             \
  } while (0)

#define MMA_SET(sa0, sa1, sb0, sb1, sb2, sb3)                            \
  do {                                                                   \
    acc[0][0] = WMMA16(sa0, sb0, acc[0][0]);                             \
    acc[0][1] = WMMA16(sa0, sb1, acc[0][1]);                             \
    acc[0][2] = WMMA16(sa0, sb2, acc[0][2]);                             \
    acc[0][3] = WMMA16(sa0, sb3, acc[0][3]);                             \
    acc[1][0] = WMMA16(sa1, sb0, acc[1][0]);                             \
    acc[1][1] = WMMA16(sa1, sb1, acc[1][1]);                             \
    acc[1][2] = WMMA16(sa1, sb2, acc[1][2]);                             \
    acc[1][3] = WMMA16(sa1, sb3, acc[1][3]);                             \
  } while (0)

__device__ __forceinline__ void gemm_core_32x64(const _Float16* __restrict__ A,
                                                const _Float16* __restrict__ Bt,
                                                int m0, int n0, int lane,
                                                v8f acc[2][4]) {
  const int l16 = lane & 15;
  const _Float16* ar0 = A + (size_t)(m0 + l16) * DM;
  const _Float16* ar1 = ar0 + (size_t)16 * DM;
  const _Float16* bp0 = Bt + (size_t)(n0 + 0  + l16) * DM;
  const _Float16* bp1 = Bt + (size_t)(n0 + 16 + l16) * DM;
  const _Float16* bp2 = Bt + (size_t)(n0 + 32 + l16) * DM;
  const _Float16* bp3 = Bt + (size_t)(n0 + 48 + l16) * DM;

  v16h xa0, xa1, xb0, xb1, xb2, xb3;   // set 0
  v16h ya0, ya1, yb0, yb1, yb2, yb3;   // set 1

  LOAD_SET(xa0, xa1, xb0, xb1, xb2, xb3, 0);

#pragma unroll 1
  for (int kb = 0; kb < DM; kb += 64) {
    // phase A: prefetch kb+32 into set1 (kb+32 <= 736 always), use set0
    LOAD_SET(ya0, ya1, yb0, yb1, yb2, yb3, kb + 32);
    MMA_SET(xa0, xa1, xb0, xb1, xb2, xb3);
    // phase B: prefetch kb+64 into set0 (dummy re-load on last iter), use set1
    const int k2 = (kb + 64 < DM) ? kb + 64 : 0;
    LOAD_SET(xa0, xa1, xb0, xb1, xb2, xb3, k2);
    MMA_SET(ya0, ya1, yb0, yb1, yb2, yb3);
  }
}

// ---- prep kernels ----------------------------------------------------------

__global__ void f32_to_f16_kernel(const float* __restrict__ src,
                                  _Float16* __restrict__ dst, int n4) {
  int i = blockIdx.x * blockDim.x + threadIdx.x;
  if (i < n4) {
    float4 v = ((const float4*)src)[i];
    v4h o = { (_Float16)v.x, (_Float16)v.y, (_Float16)v.z, (_Float16)v.w };
    ((v4h*)dst)[i] = o;
  }
}

// W is [K=768][N=768] fp32 (row-major, in->out).  Produce Wt[N][K] f16.
__global__ void transpose_w_kernel(const float* __restrict__ W,
                                   _Float16* __restrict__ Wt) {
  __shared__ float tile[32][33];
  const int tx = threadIdx.x, ty = threadIdx.y;          // 32 x 8
  const int nIn = blockIdx.x * 32 + tx;                  // column of W (n)
  const int kIn = blockIdx.y * 32 + ty;                  // row of W (k)
#pragma unroll
  for (int j = 0; j < 4; ++j)
    tile[ty + j * 8][tx] = W[(size_t)(kIn + j * 8) * DM + nIn];
  __syncthreads();
  const int kOut = blockIdx.y * 32 + tx;
  const int nOut = blockIdx.x * 32 + ty;
#pragma unroll
  for (int j = 0; j < 4; ++j)
    Wt[(size_t)(nOut + j * 8) * DM + kOut] = (_Float16)tile[tx][ty + j * 8];
}

// ---- QKV projection GEMM: out = Xh(8192x768) * W(768x768) + bias -----------
// mode 0: store [B,H,T,Dh] (Q and K);  mode 2: store [B,H,Dh,T] (V transposed)

__global__ void __launch_bounds__(256)
qkv_gemm_kernel(const _Float16* __restrict__ Xh,
                const _Float16* __restrict__ Wt,   // [N][K]
                const float*    __restrict__ bias,
                _Float16*       __restrict__ outp,
                int mode) {
  const int lane = threadIdx.x & 31;
  const int wave = threadIdx.x >> 5;
  const int m0 = blockIdx.x * 256 + wave * 32;
  const int n0 = blockIdx.y * 64;
  const int l16 = lane & 15;
  const int half = lane >> 4;

  v8f acc[2][4];
#pragma unroll
  for (int mt = 0; mt < 2; ++mt)
#pragma unroll
    for (int nt = 0; nt < 4; ++nt)
#pragma unroll
      for (int r = 0; r < 8; ++r) acc[mt][nt][r] = 0.0f;

  gemm_core_32x64(Xh, Wt, m0, n0, lane, acc);

  // C/D layout: VGPR r, lanes 0-15 -> row base+r, lanes 16-31 -> row base+8+r.
#pragma unroll
  for (int mt = 0; mt < 2; ++mt) {
    const int rbase = m0 + mt * 16 + half * 8;
#pragma unroll
    for (int nt = 0; nt < 4; ++nt) {
      const int n = n0 + nt * 16 + l16;
      const float bv = bias[n];
      const int h = n / DHEAD, d = n % DHEAD;
#pragma unroll
      for (int r = 0; r < 8; ++r) {
        const int m = rbase + r;
        const int b = m >> 10, t = m & (TSEQ - 1);
        const float v = acc[mt][nt][r] + bv;
        size_t idx;
        if (mode == 2) idx = ((size_t)(b * HEADS + h) * DHEAD + d) * TSEQ + t;
        else           idx = ((size_t)(b * HEADS + h) * TSEQ + t) * DHEAD + d;
        outp[idx] = (_Float16)v;
      }
    }
  }
}

// ---- flash attention: per wave a 32-query tile, 32-key blocks --------------

__global__ void __launch_bounds__(256)
attn_kernel(const _Float16* __restrict__ Qh,  // [B,H,T,Dh]
            const _Float16* __restrict__ Kh,  // [B,H,T,Dh]
            const _Float16* __restrict__ Vt,  // [B,H,Dh,T]
            _Float16*       __restrict__ Ah) {// [B,T,H*Dh]
  __shared__ __align__(16) _Float16 plds[8][32][32];   // per-wave P tiles (16 KB)
  const int lane = threadIdx.x & 31;
  const int wave = threadIdx.x >> 5;
  const int l16 = lane & 15;
  const int half = lane >> 4;
  const int bh = blockIdx.x;                           // 0..95
  const int qbase = blockIdx.y * 256 + wave * 32;      // 32 q rows per wave

  const _Float16* Qb = Qh + (size_t)bh * TSEQ * DHEAD;
  const _Float16* Kb = Kh + (size_t)bh * TSEQ * DHEAD;
  const _Float16* Vb = Vt + (size_t)bh * DHEAD * TSEQ;

  // Q fragments: qf[mt][kqi], mt = query sub-tile, kqi = d 0..31 / 32..63
  v16h qf[2][2];
#pragma unroll
  for (int mt = 0; mt < 2; ++mt) {
    const int qrow = qbase + mt * 16 + l16;
    qf[mt][0] = load_afrag(Qb + (size_t)qrow * DHEAD + 0,  lane);
    qf[mt][1] = load_afrag(Qb + (size_t)qrow * DHEAD + 32, lane);
  }

  float mrow[2][8], lrow[2][8];
  v8f o[2][4];
#pragma unroll
  for (int mt = 0; mt < 2; ++mt) {
#pragma unroll
    for (int r = 0; r < 8; ++r) { mrow[mt][r] = -1e30f; lrow[mt][r] = 0.0f; }
#pragma unroll
    for (int dt = 0; dt < 4; ++dt)
#pragma unroll
      for (int r = 0; r < 8; ++r) o[mt][dt][r] = 0.0f;
  }

  const int kend = qbase + 32;                 // keys 0..qbase+31 needed
#pragma unroll 1
  for (int kb = 0; kb < kend; kb += 32) {
    // K fragments for the two 16-key sub-tiles (shared by both query tiles)
    v16h bk[2][2];
#pragma unroll
    for (int st = 0; st < 2; ++st) {
      const int key = kb + st * 16 + l16;
      bk[st][0] = load_bfrag(Kb + (size_t)key * DHEAD + 0,  lane);
      bk[st][1] = load_bfrag(Kb + (size_t)key * DHEAD + 32, lane);
    }
    // issue V fragments early so they overlap the softmax VALU work below
    v16h vf[4];
#pragma unroll
    for (int dt = 0; dt < 4; ++dt)
      vf[dt] = load_bfrag(Vb + (size_t)(dt * 16 + l16) * TSEQ + kb, lane);

    // scores: s[mt][st] is a 16(q) x 16(key) tile, inner dim d = 64
    v8f s[2][2];
#pragma unroll
    for (int mt = 0; mt < 2; ++mt)
#pragma unroll
      for (int st = 0; st < 2; ++st) {
        v8f z;
#pragma unroll
        for (int r = 0; r < 8; ++r) z[r] = 0.0f;
        z = WMMA16(qf[mt][0], bk[st][0], z);
        s[mt][st] = WMMA16(qf[mt][1], bk[st][1], z);
      }

    // per query sub-tile: scale + causal mask + fused online-softmax update;
    // p values go straight to LDS, o rescale applied inline (no temp arrays)
#pragma unroll
    for (int mt = 0; mt < 2; ++mt) {
#pragma unroll
      for (int r = 0; r < 8; ++r) {
        const int q = qbase + mt * 16 + half * 8 + r;
        float v0 = s[mt][0][r] * 0.125f;       // 1/sqrt(64)
        float v1 = s[mt][1][r] * 0.125f;
        if (kb + l16 > q)      v0 = -1e30f;
        if (kb + 16 + l16 > q) v1 = -1e30f;
        float mx = fmaxf(v0, v1);
#pragma unroll
        for (int off = 8; off >= 1; off >>= 1)
          mx = fmaxf(mx, __shfl_xor(mx, off, 32));
        const float mn = fmaxf(mrow[mt][r], mx);
        const float sc = __expf(mrow[mt][r] - mn);
        const float p0 = __expf(v0 - mn);
        const float p1 = __expf(v1 - mn);
        float su = p0 + p1;
#pragma unroll
        for (int off = 8; off >= 1; off >>= 1)
          su += __shfl_xor(su, off, 32);
        lrow[mt][r] = lrow[mt][r] * sc + su;
        mrow[mt][r] = mn;
#pragma unroll
        for (int dt = 0; dt < 4; ++dt) o[mt][dt][r] *= sc;
        plds[wave][mt * 16 + half * 8 + r][l16]      = (_Float16)p0;
        plds[wave][mt * 16 + half * 8 + r][16 + l16] = (_Float16)p1;
      }
    }
    asm volatile("s_wait_dscnt 0" ::: "memory");  // DS in-order per wave; belt+braces

#pragma unroll
    for (int mt = 0; mt < 2; ++mt) {
      const v16h pf = load_afrag(&plds[wave][mt * 16 + l16][0], lane);
#pragma unroll
      for (int dt = 0; dt < 4; ++dt)
        o[mt][dt] = WMMA16(pf, vf[dt], o[mt][dt]);
    }
  }

  // normalize and store as [B, T, H*Dh] f16 for the output projection
  const int b = bh / HEADS, h = bh % HEADS;
#pragma unroll
  for (int mt = 0; mt < 2; ++mt)
#pragma unroll
    for (int r = 0; r < 8; ++r) {
      const float inv = 1.0f / lrow[mt][r];
      const int t = qbase + mt * 16 + half * 8 + r;
#pragma unroll
      for (int dt = 0; dt < 4; ++dt) {
        const int col = h * DHEAD + dt * 16 + l16;
        Ah[((size_t)(b * TSEQ + t)) * DM + col] = (_Float16)(o[mt][dt][r] * inv);
      }
    }
}

// ---- output projection: d_out(f32) = Ah * Wo + bo --------------------------

__global__ void __launch_bounds__(256)
out_gemm_kernel(const _Float16* __restrict__ Ah,
                const _Float16* __restrict__ Wto,  // [N][K]
                const float*    __restrict__ bo,
                float*          __restrict__ out) {
  const int lane = threadIdx.x & 31;
  const int wave = threadIdx.x >> 5;
  const int m0 = blockIdx.x * 256 + wave * 32;
  const int n0 = blockIdx.y * 64;
  const int l16 = lane & 15;
  const int half = lane >> 4;

  v8f acc[2][4];
#pragma unroll
  for (int mt = 0; mt < 2; ++mt)
#pragma unroll
    for (int nt = 0; nt < 4; ++nt)
#pragma unroll
      for (int r = 0; r < 8; ++r) acc[mt][nt][r] = 0.0f;

  gemm_core_32x64(Ah, Wto, m0, n0, lane, acc);

#pragma unroll
  for (int mt = 0; mt < 2; ++mt) {
    const int rbase = m0 + mt * 16 + half * 8;
#pragma unroll
    for (int nt = 0; nt < 4; ++nt) {
      const int n = n0 + nt * 16 + l16;
      const float bv = bo[n];
#pragma unroll
      for (int r = 0; r < 8; ++r)
        out[(size_t)(rbase + r) * DM + n] = acc[mt][nt][r] + bv;
    }
  }
}

// ---------------------------------------------------------------------------

extern "C" void kernel_launch(void* const* d_in, const int* in_sizes, int n_in,
                              void* d_out, int out_size, void* d_ws, size_t ws_size,
                              hipStream_t stream) {
  const float* X  = (const float*)d_in[0];
  const float* Wq = (const float*)d_in[1];
  const float* bq = (const float*)d_in[2];
  const float* Wk = (const float*)d_in[3];
  const float* bk = (const float*)d_in[4];
  const float* Wv = (const float*)d_in[5];
  const float* bv = (const float*)d_in[6];
  const float* Wo = (const float*)d_in[7];
  const float* bo = (const float*)d_in[8];
  float* out = (float*)d_out;

  // workspace partition (f16): ~68 MB total
  _Float16* p   = (_Float16*)d_ws;
  _Float16* Xh  = p; p += (size_t)MROWS * DM;
  _Float16* Wtq = p; p += (size_t)DM * DM;
  _Float16* Wtk = p; p += (size_t)DM * DM;
  _Float16* Wtv = p; p += (size_t)DM * DM;
  _Float16* Wto = p; p += (size_t)DM * DM;
  _Float16* Qh  = p; p += (size_t)MROWS * DM;
  _Float16* Kh  = p; p += (size_t)MROWS * DM;
  _Float16* Vt  = p; p += (size_t)MROWS * DM;
  _Float16* Ah  = p; p += (size_t)MROWS * DM;

  const int n4 = MROWS * DM / 4;
  f32_to_f16_kernel<<<n4 / 256, 256, 0, stream>>>(X, Xh, n4);

  dim3 tb(32, 8), tg(DM / 32, DM / 32);
  transpose_w_kernel<<<tg, tb, 0, stream>>>(Wq, Wtq);
  transpose_w_kernel<<<tg, tb, 0, stream>>>(Wk, Wtk);
  transpose_w_kernel<<<tg, tb, 0, stream>>>(Wv, Wtv);
  transpose_w_kernel<<<tg, tb, 0, stream>>>(Wo, Wto);

  dim3 gg(MROWS / 256, DM / 64);          // 32 x 12 blocks, 8 waves each
  qkv_gemm_kernel<<<gg, 256, 0, stream>>>(Xh, Wtq, bq, Qh, 0);
  qkv_gemm_kernel<<<gg, 256, 0, stream>>>(Xh, Wtk, bk, Kh, 0);
  qkv_gemm_kernel<<<gg, 256, 0, stream>>>(Xh, Wtv, bv, Vt, 2);

  dim3 ag(BATCH * HEADS, TSEQ / 256);     // 96 x 4 blocks, 8 waves each
  attn_kernel<<<ag, 256, 0, stream>>>(Qh, Kh, Vt, Ah);

  out_gemm_kernel<<<gg, 256, 0, stream>>>(Ah, Wto, bo, out);
}
